// NeighborhoodModel_74466142978784
// MI455X (gfx1250) — matrix operand order repaired and checked by reading, not compiled, not run on hardware
//
#include <hip/hip_runtime.h>
#include <cstdint>

// ---------------- problem constants ----------------
#define BDIM   8
#define VDIM   1200
#define DDIM   64
#define KHOP   2
#define HHEADS 4
#define VTILES 75          // VDIM / 16
#define KTV    38          // ceil(VDIM / 32)

typedef __attribute__((ext_vector_type(16))) _Float16 v16h;
typedef __attribute__((ext_vector_type(8)))  float    v8f;
typedef __attribute__((ext_vector_type(4)))  unsigned int tdm_u32x4;
typedef __attribute__((ext_vector_type(8)))  int          tdm_i32x8;
typedef __attribute__((ext_vector_type(4)))  int          tdm_i32x4;

#if defined(__has_builtin)
#  if __has_builtin(__builtin_amdgcn_tensor_load_to_lds)
#    define HAVE_TDM 1
#  endif
#endif
#ifndef HAVE_TDM
#  define HAVE_TDM 0
#endif

// gfx1250 hardware tanh (v_tanh_f32) if the builtin exists; else libm.
#if defined(__has_builtin)
#  if __has_builtin(__builtin_amdgcn_tanhf)
#    define FAST_TANH(x) __builtin_amdgcn_tanhf(x)
#  elif __has_builtin(__builtin_amdgcn_tanh_f32)
#    define FAST_TANH(x) __builtin_amdgcn_tanh_f32(x)
#  endif
#endif
#ifndef FAST_TANH
#  define FAST_TANH(x) tanhf(x)
#endif

// ---------------- WMMA fragment helpers (ISA 7.12.2 layouts) ----------------
// 16-bit A matrix 16x32: lane L -> row M=L&15; element e -> K per table
__device__ __forceinline__ int a_k(int lane, int e) {
  int j = e >> 1, p = e & 1, hi = lane >> 4;
  return ((j & 4) << 2) | (hi << 3) | ((j & 3) << 1) | p;
}
// 16-bit B matrix 32x16: lane L -> col N=L&15; element e -> K per table (same as A)
__device__ __forceinline__ int b_k(int lane, int e) {
  int j = e >> 1, p = e & 1, hi = lane >> 4;
  return ((j & 4) << 2) | (hi << 3) | ((j & 3) << 1) | p;
}
__device__ __forceinline__ v8f vzero8() {
  v8f z = {0.f, 0.f, 0.f, 0.f, 0.f, 0.f, 0.f, 0.f};
  return z;
}
__device__ __forceinline__ v8f wmma_f16f32(v16h a, v16h b, v8f c) {
  return __builtin_amdgcn_wmma_f32_16x16x32_f16(false, a, false, b, (short)0, c, false, false);
}
// Build an A fragment from a row-major f32 LDS tile (16 x ld), K window at kbase
__device__ __forceinline__ v16h afrag_lds(const float* t, int ld, int kbase, int lane) {
  v16h a;
  int m = lane & 15;
#pragma unroll
  for (int e = 0; e < 16; ++e) a[e] = (_Float16)t[m * ld + kbase + a_k(lane, e)];
  return a;
}
// Load a pre-packed B fragment (packed layout [kt][nt][lane][16])
__device__ __forceinline__ v16h load_bfrag(const _Float16* wp, int kt, int nt, int NT, int lane) {
  return *(const v16h*)(wp + ((((size_t)kt * NT + nt) * 32 + lane) << 4));
}
// sigmoid(x) = 0.5*tanh(x/2)+0.5 -> single v_tanh_f32 on gfx1250
__device__ __forceinline__ float sigmoidf_(float x) { return 0.5f * FAST_TANH(0.5f * x) + 0.5f; }

// ---------------- Tensor Data Mover: 2D f32 tile -> LDS ----------------
// D# per CDNA5 ISA. tile (tile_d0 cols x tile_d1 rows),
// tensor_d0 = remaining row width (OOB columns read as zero), stride0 in elements.
__device__ __forceinline__ void tdm_load_tile_f32(unsigned lds_off, const void* gsrc,
                                                  unsigned tensor_d0, unsigned tensor_d1,
                                                  unsigned tile_d0, unsigned tile_d1,
                                                  unsigned long long stride0) {
#if HAVE_TDM
  unsigned long long ga = (unsigned long long)(uintptr_t)gsrc;
  tdm_u32x4 g0;
  g0[0] = 1u;                                                   // count=1, user mode
  g0[1] = lds_off;                                              // lds_addr (bytes)
  g0[2] = (unsigned)(ga & 0xffffffffu);                         // global_addr[31:0]
  g0[3] = (unsigned)((ga >> 32) & 0x01ffffffu) | 0x80000000u;   // global_addr[56:32] | type=2
  tdm_i32x8 g1;
  g1[0] = (int)(2u << 16);                                      // data_size=4B, wg_mask=0
  g1[1] = (int)((tensor_d0 & 0xffffu) << 16);                   // tensor_dim0 lo
  g1[2] = (int)((tensor_d0 >> 16) | ((tensor_d1 & 0xffffu) << 16));
  g1[3] = (int)((tensor_d1 >> 16) | (tile_d0 << 16));           // tile_dim0
  g1[4] = (int)(tile_d1 & 0xffffu);                             // tile_dim1, tile_dim2=0
  g1[5] = (int)(unsigned)(stride0 & 0xffffffffull);             // tensor_dim0_stride lo
  g1[6] = (int)(unsigned)((stride0 >> 32) & 0xffffull);         // stride hi, dim1_stride=0
  g1[7] = 0;
  tdm_i32x4 gz = {0, 0, 0, 0};
#if defined(__clang_major__) && (__clang_major__ >= 23)
  tdm_i32x8 gz8 = {0, 0, 0, 0, 0, 0, 0, 0};
  __builtin_amdgcn_tensor_load_to_lds(g0, g1, gz, gz, gz8, 0);
#else
  __builtin_amdgcn_tensor_load_to_lds(g0, g1, gz, gz, 0);
#endif
#else
  (void)lds_off; (void)gsrc; (void)tensor_d0; (void)tensor_d1;
  (void)tile_d0; (void)tile_d1; (void)stride0;
#endif
}

// ---------------- weight / activation packing ----------------
// Pack f32 KxN matrix into f16 B-fragments [kt][nt][lane][16], zero padded.
__global__ void k_pack_w(const float* __restrict__ W, _Float16* __restrict__ out,
                         int Kdim, int N, int KT, int NT) {
  int idx = blockIdx.x * 256 + threadIdx.x;
  int total = KT * NT * 512;
  if (idx >= total) return;
  int e = idx & 15, lane = (idx >> 4) & 31;
  int rest = idx >> 9;
  int nt = rest % NT, kt = rest / NT;
  int k = kt * 32 + b_k(lane, e);
  int n = nt * 16 + (lane & 15);
  float v = (k < Kdim && n < N) ? W[k * N + n] : 0.f;
  out[idx] = (_Float16)v;
}

// Pack enc [B,V,64] into per-graph B-fragments over K=V (padded to KTV*32).
__global__ void k_pack_enc(const float* __restrict__ enc, _Float16* __restrict__ encB) {
  const int perB = KTV * 4 * 512;
  int idx = blockIdx.x * 256 + threadIdx.x;
  if (idx >= BDIM * perB) return;
  int e = idx & 15, lane = (idx >> 4) & 31;
  int nt = (idx >> 9) & 3;
  int kt = (idx >> 11) % KTV;
  int b = idx / perB;
  int v = kt * 32 + b_k(lane, e);
  int n = nt * 16 + (lane & 15);
  float val = (v < VDIM) ? enc[((size_t)b * VDIM + v) * DDIM + n] : 0.f;
  encB[idx] = (_Float16)val;
}

// ---------------- degree normalization ----------------
__global__ void k_invdeg(const float* __restrict__ neigh, float* __restrict__ invdeg) {
  int warp = threadIdx.x >> 5, lane = threadIdx.x & 31;
  int row = blockIdx.x * 8 + warp;
  if (row >= KHOP * BDIM * VDIM) return;
  const float* p = neigh + (size_t)row * VDIM;
  float s = 0.f;
  for (int v = lane; v < VDIM; v += 32) s += p[v];
#pragma unroll
  for (int off = 16; off >= 1; off >>= 1) s += __shfl_xor(s, off);
  if (lane == 0) invdeg[row] = 1.f / fmaxf(s, 1.f);
}

// ---------------- encoder MLP: enc = tanh(tanh(x@W0+b0)@W1+b1) ----------------
__global__ __launch_bounds__(32) void k_encode(
    const float* __restrict__ feat, const float* __restrict__ emb,
    const _Float16* __restrict__ w0p, const float* __restrict__ b0,
    const _Float16* __restrict__ w1p, const float* __restrict__ b1,
    float* __restrict__ enc) {
  __shared__ float h[16 * 64];
  int lane = threadIdx.x;
  int m = lane & 15, mh = (lane >> 4) * 8;
  int node0 = blockIdx.x * 16;
  int node = node0 + m;
  v16h a;
#pragma unroll
  for (int e = 0; e < 16; ++e) {
    int k = a_k(lane, e);
    float v;
    if (k < 16)      v = emb[node * 16 + k];          // embeddings first (concat order)
    else if (k < 24) v = feat[node * 8 + (k - 16)];
    else             v = 0.f;
    a[e] = (_Float16)v;
  }
  v8f acc[4];
#pragma unroll
  for (int nt = 0; nt < 4; ++nt) acc[nt] = wmma_f16f32(a, load_bfrag(w0p, 0, nt, 4, lane), vzero8());
#pragma unroll
  for (int nt = 0; nt < 4; ++nt) {
    int n = (lane & 15) + nt * 16;
#pragma unroll
    for (int r = 0; r < 8; ++r) h[(mh + r) * 64 + n] = FAST_TANH(acc[nt][r] + b0[n]);
  }
  __syncthreads();
  v8f acc2[4];
#pragma unroll
  for (int nt = 0; nt < 4; ++nt) acc2[nt] = vzero8();
#pragma unroll
  for (int kt = 0; kt < 2; ++kt) {
    v16h a2 = afrag_lds(h, 64, kt * 32, lane);
#pragma unroll
    for (int nt = 0; nt < 4; ++nt)
      acc2[nt] = wmma_f16f32(a2, load_bfrag(w1p, kt, nt, 4, lane), acc2[nt]);
  }
#pragma unroll
  for (int nt = 0; nt < 4; ++nt) {
    int n = (lane & 15) + nt * 16;
#pragma unroll
    for (int r = 0; r < 8; ++r)
      enc[(size_t)(node0 + mh + r) * DDIM + n] = FAST_TANH(acc2[nt][r] + b1[n]);
  }
}

// ---------------- fused: msg = nb_norm@enc ; agg = tanh(msg@Wk+bk) ;
// ----------------        proj/scores for all heads ----------------
__global__ __launch_bounds__(32) void k_layer_agg(
    const float* __restrict__ neigh, const float* __restrict__ invdeg,
    const _Float16* __restrict__ encB,
    const _Float16* __restrict__ aggWp, const float* __restrict__ aggB,
    const _Float16* __restrict__ attWp, const float* __restrict__ attB,
    const float* __restrict__ attV,
    float* __restrict__ agg, float* __restrict__ scores) {
  __shared__ float nbt[2][16 * 32];     // double-buffered TDM destination
  __shared__ float mt[16 * 64];
  int lane = threadIdx.x;
  int m = lane & 15, mh = (lane >> 4) * 8;
  int t = blockIdx.x % VTILES;
  int b = (blockIdx.x / VTILES) % BDIM;
  int kk = blockIdx.x / (VTILES * BDIM);
  int row0 = t * 16;
  float sm = invdeg[(kk * BDIM + b) * VDIM + row0 + m];
  const float* nbbase = neigh + ((size_t)(kk * BDIM + b) * VDIM + row0) * VDIM;
  const _Float16* encBb = encB + (size_t)b * (KTV * 4 * 512);
  v8f acc[4];
#pragma unroll
  for (int nt = 0; nt < 4; ++nt) acc[nt] = vzero8();

#if HAVE_TDM
  unsigned ldsbuf[2];
  ldsbuf[0] = (unsigned)(uintptr_t)(void*)&nbt[0][0];
  ldsbuf[1] = (unsigned)(uintptr_t)(void*)&nbt[1][0];
  // software pipeline: prefetch tile kt+1 with the TDM while tile kt feeds WMMA.
  tdm_load_tile_f32(ldsbuf[0], nbbase, (unsigned)VDIM, (unsigned)VDIM, 32u, 16u,
                    (unsigned long long)VDIM);
  for (int kt = 0; kt < KTV; ++kt) {
    if (kt + 1 < KTV) {
      unsigned rem = (unsigned)(VDIM - (kt + 1) * 32);   // OOB cols -> zero fill
      tdm_load_tile_f32(ldsbuf[(kt + 1) & 1], nbbase + (kt + 1) * 32, rem, (unsigned)VDIM,
                        32u, 16u, (unsigned long long)VDIM);
      __builtin_amdgcn_s_wait_tensorcnt(1);   // oldest of the two loads has landed
    } else {
      __builtin_amdgcn_s_wait_tensorcnt(0);
    }
    const float* tile = &nbt[kt & 1][0];
    v16h a;
#pragma unroll
    for (int e = 0; e < 16; ++e) a[e] = (_Float16)(tile[m * 32 + a_k(lane, e)] * sm);
#pragma unroll
    for (int nt = 0; nt < 4; ++nt)
      acc[nt] = wmma_f16f32(
          a, *(const v16h*)(encBb + (((size_t)kt * 4 + nt) * 32 + lane) * 16), acc[nt]);
    // single wave: the compiler's dscnt waits before the WMMAs guarantee these LDS
    // reads complete before the TDM that overwrites this buffer (issued next iteration).
  }
#else
  for (int kt = 0; kt < KTV; ++kt) {
    unsigned rem = (unsigned)(VDIM - kt * 32);
#pragma unroll
    for (int i = 0; i < 16; ++i) {
      int idx = lane + i * 32;
      int r = idx >> 5, c = idx & 31;
      nbt[0][idx] = ((unsigned)c < rem) ? nbbase[(size_t)r * VDIM + kt * 32 + c] : 0.f;
    }
    __syncthreads();
    v16h a;
#pragma unroll
    for (int e = 0; e < 16; ++e) a[e] = (_Float16)(nbt[0][m * 32 + a_k(lane, e)] * sm);
#pragma unroll
    for (int nt = 0; nt < 4; ++nt)
      acc[nt] = wmma_f16f32(
          a, *(const v16h*)(encBb + (((size_t)kt * 4 + nt) * 32 + lane) * 16), acc[nt]);
    __syncthreads();
  }
#endif
  // msg tile -> LDS
#pragma unroll
  for (int nt = 0; nt < 4; ++nt) {
    int n = (lane & 15) + nt * 16;
#pragma unroll
    for (int r = 0; r < 8; ++r) mt[(mh + r) * 64 + n] = acc[nt][r];
  }
  __syncthreads();
  // agg = tanh(msg @ aggW[kk] + aggB[kk])
  const _Float16* wp = aggWp + (size_t)kk * 4096;
  v8f ag[4];
#pragma unroll
  for (int nt = 0; nt < 4; ++nt) ag[nt] = vzero8();
#pragma unroll
  for (int kt2 = 0; kt2 < 2; ++kt2) {
    v16h a2 = afrag_lds(mt, 64, kt2 * 32, lane);
#pragma unroll
    for (int nt = 0; nt < 4; ++nt)
      ag[nt] = wmma_f16f32(a2, load_bfrag(wp, kt2, nt, 4, lane), ag[nt]);
  }
  __syncthreads();
#pragma unroll
  for (int nt = 0; nt < 4; ++nt) {
    int n = (lane & 15) + nt * 16;
#pragma unroll
    for (int r = 0; r < 8; ++r) {
      float x = FAST_TANH(ag[nt][r] + aggB[kk * DDIM + n]);
      mt[(mh + r) * 64 + n] = x;
      agg[((((size_t)kk * BDIM + b) * VDIM) + row0 + mh + r) * DDIM + n] = x;
    }
  }
  __syncthreads();
  // attention projections + scores per head (proj=tanh(agg@attW+attB), score=proj.attV)
  v16h pa0 = afrag_lds(mt, 64, 0, lane);
  v16h pa1 = afrag_lds(mt, 64, 32, lane);
  for (int h = 0; h < HHEADS; ++h) {
    const _Float16* awp = attWp + (size_t)h * 4096;
    v8f pr[4];
#pragma unroll
    for (int nt = 0; nt < 4; ++nt) {
      pr[nt] = wmma_f16f32(pa0, load_bfrag(awp, 0, nt, 4, lane), vzero8());
      pr[nt] = wmma_f16f32(pa1, load_bfrag(awp, 1, nt, 4, lane), pr[nt]);
    }
    float part[8] = {0.f, 0.f, 0.f, 0.f, 0.f, 0.f, 0.f, 0.f};
#pragma unroll
    for (int nt = 0; nt < 4; ++nt) {
      int n = (lane & 15) + nt * 16;
      float av = attV[h * DDIM + n], ab = attB[h * DDIM + n];
#pragma unroll
      for (int r = 0; r < 8; ++r) part[r] += FAST_TANH(pr[nt][r] + ab) * av;
    }
#pragma unroll
    for (int off = 8; off >= 1; off >>= 1)
#pragma unroll
      for (int r = 0; r < 8; ++r) part[r] += __shfl_xor(part[r], off);
    if ((lane & 15) == 0) {
#pragma unroll
      for (int r = 0; r < 8; ++r)
        scores[((((size_t)h * KHOP + kk) * BDIM) + b) * VDIM + row0 + mh + r] = part[r];
    }
  }
}

// ---------------- softmax-over-hops combine + GRU update (in-place enc) ----------------
__global__ __launch_bounds__(32) void k_attn_gru(
    const float* __restrict__ scores, const float* __restrict__ agg,
    const _Float16* __restrict__ wup, const float* __restrict__ bu,
    const _Float16* __restrict__ wrp, const float* __restrict__ br,
    const _Float16* __restrict__ wcp, const float* __restrict__ bc,
    float* __restrict__ enc) {
  __shared__ float xs[16 * 128];
  __shared__ float et[16 * 64];
  __shared__ float ut[16 * 64];
  __shared__ float c0[16], c1[16];
  int lane = threadIdx.x;
  int mh = (lane >> 4) * 8;
  int t = blockIdx.x % VTILES, b = blockIdx.x / VTILES;
  int row0 = t * 16;
  if (lane < 16) {
    float a0 = 0.f, a1 = 0.f;
    int v = row0 + lane;
#pragma unroll
    for (int h = 0; h < HHEADS; ++h) {
      float s0 = scores[((((size_t)h * KHOP + 0) * BDIM) + b) * VDIM + v];
      float s1 = scores[((((size_t)h * KHOP + 1) * BDIM) + b) * VDIM + v];
      // softmax over 2 hops == logistic of the difference (one v_tanh_f32)
      float w0 = sigmoidf_(s0 - s1);
      a0 += w0;
      a1 += 1.f - w0;
    }
    c0[lane] = a0 * 0.25f;   // /H
    c1[lane] = a1 * 0.25f;
  }
  __syncthreads();
  const float* encT = enc + (size_t)(b * VDIM + row0) * DDIM;
  const float* ag0 = agg + (((size_t)0 * BDIM + b) * VDIM + row0) * DDIM;
  const float* ag1 = agg + (((size_t)1 * BDIM + b) * VDIM + row0) * DDIM;
#pragma unroll
  for (int i = 0; i < 32; ++i) {
    int idx = lane + i * 32;
    int mm = idx >> 6, n = idx & 63;
    float ev = encT[idx];
    et[idx] = ev;
    xs[mm * 128 + 64 + n] = ev;
    xs[mm * 128 + n] = c0[mm] * ag0[idx] + c1[mm] * ag1[idx];
  }
  __syncthreads();
  v8f au[4], ar[4];
#pragma unroll
  for (int nt = 0; nt < 4; ++nt) { au[nt] = vzero8(); ar[nt] = vzero8(); }
#pragma unroll
  for (int kt = 0; kt < 4; ++kt) {
    v16h xa = afrag_lds(xs, 128, kt * 32, lane);
#pragma unroll
    for (int nt = 0; nt < 4; ++nt) {
      au[nt] = wmma_f16f32(xa, load_bfrag(wup, kt, nt, 4, lane), au[nt]);
      ar[nt] = wmma_f16f32(xa, load_bfrag(wrp, kt, nt, 4, lane), ar[nt]);
    }
  }
  __syncthreads();
#pragma unroll
  for (int nt = 0; nt < 4; ++nt) {
    int n = (lane & 15) + nt * 16;
#pragma unroll
    for (int r = 0; r < 8; ++r) {
      int mm = mh + r;
      ut[mm * 64 + n] = sigmoidf_(au[nt][r] + bu[n]);
      float rr = sigmoidf_(ar[nt][r] + br[n]);
      xs[mm * 128 + 64 + n] = rr * et[mm * 64 + n];   // [nxt | r*enc]
    }
  }
  __syncthreads();
  v8f ac[4];
#pragma unroll
  for (int nt = 0; nt < 4; ++nt) ac[nt] = vzero8();
#pragma unroll
  for (int kt = 0; kt < 4; ++kt) {
    v16h xa = afrag_lds(xs, 128, kt * 32, lane);
#pragma unroll
    for (int nt = 0; nt < 4; ++nt)
      ac[nt] = wmma_f16f32(xa, load_bfrag(wcp, kt, nt, 4, lane), ac[nt]);
  }
#pragma unroll
  for (int nt = 0; nt < 4; ++nt) {
    int n = (lane & 15) + nt * 16;
#pragma unroll
    for (int r = 0; r < 8; ++r) {
      int mm = mh + r;
      float cc = FAST_TANH(ac[nt][r] + bc[n]);
      float uu = ut[mm * 64 + n];
      float ev = et[mm * 64 + n];
      enc[(size_t)(b * VDIM + row0 + mm) * DDIM + n] = uu * ev + (1.f - uu) * cc;
    }
  }
}

// ---------------- linear decoders: pred and dual potentials ----------------
__global__ __launch_bounds__(32) void k_decode(
    const float* __restrict__ enc,
    const _Float16* __restrict__ dw0p, const float* __restrict__ db0,
    const _Float16* __restrict__ dw1p, const float* __restrict__ db1,
    const _Float16* __restrict__ uw0p, const float* __restrict__ ub0,
    const _Float16* __restrict__ uw1p, const float* __restrict__ ub1,
    float* __restrict__ pred, float* __restrict__ dualp) {
  __shared__ float et[16 * 64];
  __shared__ float ht[16 * 64];
  int lane = threadIdx.x;
  int mh = (lane >> 4) * 8;
  int t = blockIdx.x % VTILES, b = blockIdx.x / VTILES;
  int row0 = t * 16;
  const float* encT = enc + (size_t)(b * VDIM + row0) * DDIM;
#pragma unroll
  for (int i = 0; i < 32; ++i) et[lane + i * 32] = encT[lane + i * 32];
  __syncthreads();
  v16h ea0 = afrag_lds(et, 64, 0, lane);
  v16h ea1 = afrag_lds(et, 64, 32, lane);
  for (int path = 0; path < 2; ++path) {
    const _Float16* w0p_ = path ? uw0p : dw0p;
    const float* b0_ = path ? ub0 : db0;
    const _Float16* w1p_ = path ? uw1p : dw1p;
    const float* b1_ = path ? ub1 : db1;
    float* o = path ? dualp : pred;
    v8f hh[4];
#pragma unroll
    for (int nt = 0; nt < 4; ++nt) {
      hh[nt] = wmma_f16f32(ea0, load_bfrag(w0p_, 0, nt, 4, lane), vzero8());
      hh[nt] = wmma_f16f32(ea1, load_bfrag(w0p_, 1, nt, 4, lane), hh[nt]);
    }
    __syncthreads();
#pragma unroll
    for (int nt = 0; nt < 4; ++nt) {
      int n = (lane & 15) + nt * 16;
#pragma unroll
      for (int r = 0; r < 8; ++r) ht[(mh + r) * 64 + n] = hh[nt][r] + b0_[n];
    }
    __syncthreads();
    v8f pp = wmma_f16f32(afrag_lds(ht, 64, 0, lane), load_bfrag(w1p_, 0, 0, 1, lane), vzero8());
    pp = wmma_f16f32(afrag_lds(ht, 64, 32, lane), load_bfrag(w1p_, 1, 0, 1, lane), pp);
    if ((lane & 15) == 0) {
#pragma unroll
      for (int r = 0; r < 8; ++r) o[(size_t)b * VDIM + row0 + mh + r] = pp[r] + b1_[0];
    }
    __syncthreads();
  }
}

// ---------------- sparsemax per adjacency row (bisection on tau) ----------------
__global__ void k_sparsemax(const float* __restrict__ adj, const float* __restrict__ pred,
                            float* __restrict__ props, float* __restrict__ rowsq) {
  int warp = threadIdx.x >> 5, lane = threadIdx.x & 31;
  int row = blockIdx.x * 8 + warp;
  if (row >= BDIM * VDIM) return;
  int b = row / VDIM;
  const float* arow = adj + (size_t)row * VDIM;
  const float* pb = pred + (size_t)b * VDIM;
  float z[KTV];
  float mx = -3.0e38f;
  int cnt = 0;
#pragma unroll
  for (int i = 0; i < KTV; ++i) {
    int v = lane + i * 32;
    float a = (v < VDIM) ? arow[v] : 0.f;
    float zz = (a > 0.f) ? pb[v] : -3.0e38f;
    z[i] = zz;
    if (a > 0.f) { cnt++; mx = fmaxf(mx, zz); }
  }
#pragma unroll
  for (int off = 16; off >= 1; off >>= 1) {
    mx = fmaxf(mx, __shfl_xor(mx, off));
    cnt += __shfl_xor(cnt, off);
  }
  float lo = mx - 1.f, hi = mx;     // sum relu(z - (mx-1)) >= 1, sum relu(z - mx) == 0
  for (int it = 0; it < 30; ++it) {
    float tau = 0.5f * (lo + hi);
    float s = 0.f;
#pragma unroll
    for (int i = 0; i < KTV; ++i) s += fmaxf(z[i] - tau, 0.f);
#pragma unroll
    for (int off = 16; off >= 1; off >>= 1) s += __shfl_xor(s, off);
    if (s > 1.f) lo = tau; else hi = tau;
  }
  float tau = 0.5f * (lo + hi);
  float sq = 0.f;
#pragma unroll
  for (int i = 0; i < KTV; ++i) {
    int v = lane + i * 32;
    if (v < VDIM) {
      float p = (cnt > 0) ? fmaxf(z[i] - tau, 0.f) : 0.f;
      props[(size_t)row * VDIM + v] = p;
      sq += p * p;
    }
  }
#pragma unroll
  for (int off = 16; off >= 1; off >>= 1) sq += __shfl_xor(sq, off);
  if (lane == 0) rowsq[row] = sq;
}

// ---------------- min-cost-flow excess iterations ----------------
__global__ void k_flow_init(const float* __restrict__ demand, float* __restrict__ ex,
                            float* __restrict__ edge_acc) {
  int i = blockIdx.x * 256 + threadIdx.x;
  if (i < BDIM * VDIM) ex[i] = fmaxf(-demand[i], 0.f);
  if (i < BDIM) edge_acc[i] = 0.f;
}

__global__ void k_flow_iter(const float* __restrict__ props, const float* __restrict__ exIn,
                            const float* __restrict__ demand, float* __restrict__ exOut) {
  __shared__ float exs[VDIM];
  int b = blockIdx.y;
  int v = blockIdx.x * 256 + threadIdx.x;
  for (int i = threadIdx.x; i < VDIM; i += 256) exs[i] = exIn[(size_t)b * VDIM + i];
  __syncthreads();
  if (v >= VDIM) return;
  const float* pb = props + (size_t)b * VDIM * VDIM + v;
  float acc = 0.f;
  for (int u = 0; u < VDIM; ++u) {
    if ((u & 63) == 0) __builtin_prefetch(pb + (size_t)(u + 64) * VDIM, 0, 1);
    acc = fmaf(pb[(size_t)u * VDIM], exs[u], acc);
  }
  exOut[(size_t)b * VDIM + v] = fmaxf(acc - demand[(size_t)b * VDIM + v], 0.f);
}

// ---------------- reductions: flow cost, dual demand ----------------
__global__ void k_costs(const float* __restrict__ ex, const float* __restrict__ rowsq,
                        const float* __restrict__ dualp, const float* __restrict__ demand,
                        float* __restrict__ flowc, float* __restrict__ dualdem) {
  __shared__ float r1[256], r2[256];
  int b = blockIdx.x, tid = threadIdx.x;
  float s1 = 0.f, s2 = 0.f;
  for (int v = tid; v < VDIM; v += 256) {
    float e = ex[(size_t)b * VDIM + v];
    s1 += e * e * rowsq[(size_t)b * VDIM + v];
    s2 += dualp[(size_t)b * VDIM + v] * demand[(size_t)b * VDIM + v];
  }
  r1[tid] = s1; r2[tid] = s2;
  __syncthreads();
  for (int s = 128; s > 0; s >>= 1) {
    if (tid < s) { r1[tid] += r1[tid + s]; r2[tid] += r2[tid + s]; }
    __syncthreads();
  }
  if (tid == 0) { flowc[b] = r1[0]; dualdem[b] = r2[0]; }
}

// ---------------- per-edge Nesterov dual-flow recurrence + cost ----------------
__global__ void k_dual_edges(const float* __restrict__ adj, const float* __restrict__ dualp,
                             float* __restrict__ edge_acc) {
  __shared__ float red[256];
  int b = blockIdx.y;
  const float* ab = adj + (size_t)b * VDIM * VDIM;
  const float* db = dualp + (size_t)b * VDIM;
  float local = 0.f;
  size_t total = (size_t)VDIM * VDIM;
  for (size_t idx = (size_t)blockIdx.x * 256 + threadIdx.x; idx < total;
       idx += (size_t)256 * gridDim.x) {
    float a = ab[idx];
    if (a != 0.f) {
      int u = (int)(idx / VDIM), v = (int)(idx % VDIM);
      float dd = db[u] - db[v];
      float f = 0.f, acm = 0.f;
#pragma unroll
      for (int it = 0; it < 8; ++it) {
        float look = f - 0.9f * acm;
        float g = 2.f * look - dd;
        acm = 0.9f * acm + 0.1f * g;
        f = fmaxf(f - acm, 0.f);
      }
      local += f * f - dd * f;
    }
  }
  red[threadIdx.x] = local;
  __syncthreads();
  for (int s = 128; s > 0; s >>= 1) {
    if (threadIdx.x < s) red[threadIdx.x] += red[threadIdx.x + s];
    __syncthreads();
  }
  if (threadIdx.x == 0) atomicAdd(&edge_acc[b], red[0]);
}

__global__ void k_final(const float* __restrict__ flowc, const float* __restrict__ edge_acc,
                        const float* __restrict__ dualdem, float* __restrict__ out) {
  int i = threadIdx.x;
  if (i < BDIM) out[i] = flowc[i] - edge_acc[i] + dualdem[i];
}

// ---------------- host orchestration ----------------
extern "C" void kernel_launch(void* const* d_in, const int* in_sizes, int n_in,
                              void* d_out, int out_size, void* d_ws, size_t ws_size,
                              hipStream_t stream) {
  (void)in_sizes; (void)n_in; (void)out_size; (void)ws_size;
  const float* feat    = (const float*)d_in[0];
  const float* emb     = (const float*)d_in[1];
  const float* demand  = (const float*)d_in[2];
  const float* adj     = (const float*)d_in[3];
  const float* neigh   = (const float*)d_in[4];
  const float* enc_w0  = (const float*)d_in[5];
  const float* enc_b0  = (const float*)d_in[6];
  const float* enc_w1  = (const float*)d_in[7];
  const float* enc_b1  = (const float*)d_in[8];
  const float* agg_w   = (const float*)d_in[9];
  const float* agg_b   = (const float*)d_in[10];
  const float* att_w   = (const float*)d_in[11];
  const float* att_b   = (const float*)d_in[12];
  const float* att_v   = (const float*)d_in[13];
  const float* gru_wu  = (const float*)d_in[14];
  const float* gru_bu  = (const float*)d_in[15];
  const float* gru_wr  = (const float*)d_in[16];
  const float* gru_br  = (const float*)d_in[17];
  const float* gru_wc  = (const float*)d_in[18];
  const float* gru_bc  = (const float*)d_in[19];
  const float* dec_w0  = (const float*)d_in[20];
  const float* dec_b0  = (const float*)d_in[21];
  const float* dec_w1  = (const float*)d_in[22];
  const float* dec_b1  = (const float*)d_in[23];
  const float* dual_w0 = (const float*)d_in[24];
  const float* dual_b0 = (const float*)d_in[25];
  const float* dual_w1 = (const float*)d_in[26];
  const float* dual_b1 = (const float*)d_in[27];
  float* out = (float*)d_out;

  char* ws = (char*)d_ws;
  size_t off = 0;
  auto alloc = [&](size_t bytes) -> void* {
    void* p = ws + off;
    off = (off + bytes + 255) & ~(size_t)255;
    return p;
  };
  float* enc      = (float*)alloc(sizeof(float) * BDIM * VDIM * DDIM);
  _Float16* encB  = (_Float16*)alloc(sizeof(_Float16) * (size_t)BDIM * KTV * 4 * 512);
  float* invdeg   = (float*)alloc(sizeof(float) * KHOP * BDIM * VDIM);
  float* agg      = (float*)alloc(sizeof(float) * (size_t)KHOP * BDIM * VDIM * DDIM);
  float* scores   = (float*)alloc(sizeof(float) * HHEADS * KHOP * BDIM * VDIM);
  float* pred     = (float*)alloc(sizeof(float) * BDIM * VDIM);
  float* dualp    = (float*)alloc(sizeof(float) * BDIM * VDIM);
  float* props    = (float*)alloc(sizeof(float) * (size_t)BDIM * VDIM * VDIM);
  float* rowsq    = (float*)alloc(sizeof(float) * BDIM * VDIM);
  float* exA      = (float*)alloc(sizeof(float) * BDIM * VDIM);
  float* exB      = (float*)alloc(sizeof(float) * BDIM * VDIM);
  float* flowc    = (float*)alloc(sizeof(float) * BDIM);
  float* dualdem  = (float*)alloc(sizeof(float) * BDIM);
  float* edgeacc  = (float*)alloc(sizeof(float) * BDIM);
  _Float16* w0p   = (_Float16*)alloc(sizeof(_Float16) * 2048);
  _Float16* w1p   = (_Float16*)alloc(sizeof(_Float16) * 4096);
  _Float16* aggp  = (_Float16*)alloc(sizeof(_Float16) * 2 * 4096);
  _Float16* attp  = (_Float16*)alloc(sizeof(_Float16) * 4 * 4096);
  _Float16* wup   = (_Float16*)alloc(sizeof(_Float16) * 8192);
  _Float16* wrp   = (_Float16*)alloc(sizeof(_Float16) * 8192);
  _Float16* wcp   = (_Float16*)alloc(sizeof(_Float16) * 8192);
  _Float16* dw0p  = (_Float16*)alloc(sizeof(_Float16) * 4096);
  _Float16* dw1p  = (_Float16*)alloc(sizeof(_Float16) * 1024);
  _Float16* uw0p  = (_Float16*)alloc(sizeof(_Float16) * 4096);
  _Float16* uw1p  = (_Float16*)alloc(sizeof(_Float16) * 1024);

  auto packw = [&](const float* W, _Float16* o, int K, int N, int KT, int NT) {
    int total = KT * NT * 512;
    k_pack_w<<<(total + 255) / 256, 256, 0, stream>>>(W, o, K, N, KT, NT);
  };
  packw(enc_w0, w0p, 24, 64, 1, 4);
  packw(enc_w1, w1p, 64, 64, 2, 4);
  for (int k = 0; k < KHOP; ++k) packw(agg_w + (size_t)k * 4096, aggp + (size_t)k * 4096, 64, 64, 2, 4);
  for (int h = 0; h < HHEADS; ++h) packw(att_w + (size_t)h * 4096, attp + (size_t)h * 4096, 64, 64, 2, 4);
  packw(gru_wu, wup, 128, 64, 4, 4);
  packw(gru_wr, wrp, 128, 64, 4, 4);
  packw(gru_wc, wcp, 128, 64, 4, 4);
  packw(dec_w0, dw0p, 64, 64, 2, 4);
  packw(dec_w1, dw1p, 64, 1, 2, 1);
  packw(dual_w0, uw0p, 64, 64, 2, 4);
  packw(dual_w1, uw1p, 64, 1, 2, 1);

  k_invdeg<<<(KHOP * BDIM * VDIM) / 8, 256, 0, stream>>>(neigh, invdeg);
  k_encode<<<BDIM * VDIM / 16, 32, 0, stream>>>(feat, emb, w0p, enc_b0, w1p, enc_b1, enc);

  const int packEncTotal = BDIM * KTV * 4 * 512;
  k_pack_enc<<<(packEncTotal + 255) / 256, 256, 0, stream>>>(enc, encB);

  for (int layer = 0; layer < 2; ++layer) {
    k_layer_agg<<<KHOP * BDIM * VTILES, 32, 0, stream>>>(
        neigh, invdeg, encB, aggp, agg_b, attp, att_b, att_v, agg, scores);
    k_attn_gru<<<BDIM * VTILES, 32, 0, stream>>>(
        scores, agg, wup, gru_bu, wrp, gru_br, wcp, gru_bc, enc);
    if (layer == 0)
      k_pack_enc<<<(packEncTotal + 255) / 256, 256, 0, stream>>>(enc, encB);
  }

  k_decode<<<BDIM * VTILES, 32, 0, stream>>>(enc, dw0p, dec_b0, dw1p, dec_b1,
                                             uw0p, dual_b0, uw1p, dual_b1, pred, dualp);

  k_sparsemax<<<(BDIM * VDIM) / 8, 256, 0, stream>>>(adj, pred, props, rowsq);
  k_flow_init<<<(BDIM * VDIM + 255) / 256, 256, 0, stream>>>(demand, exA, edgeacc);

  float* ei = exA;
  float* eo = exB;
  for (int it = 0; it < 7; ++it) {   // FLOW_ITERS=8 total: init + 7 matvec relaxations
    k_flow_iter<<<dim3((VDIM + 255) / 256, BDIM), 256, 0, stream>>>(props, ei, demand, eo);
    float* tmp = ei; ei = eo; eo = tmp;
  }

  k_costs<<<BDIM, 256, 0, stream>>>(ei, rowsq, dualp, demand, flowc, dualdem);
  k_dual_edges<<<dim3(64, BDIM), 256, 0, stream>>>(adj, dualp, edgeacc);
  k_final<<<1, 32, 0, stream>>>(flowc, edgeacc, dualdem, out);
}
